// udkws_aatest_29910152249986
// MI455X (gfx1250) — compile-verified, matrix operand order checked by reading, not run on hardware
//
#include <hip/hip_runtime.h>
#include <hip/hip_bf16.h>

// ---------------------------------------------------------------------------
// CDNA5 / gfx1250 implementation.
// Input flattening assumption (setup_inputs() dict insertion order, params
// recursed in insertion order):
//  0 anco_speech f32[128,100,384]   1 anchor_mask i32[128,100]
//  2 com_speech  f32[128,2000,384]  3 comparison_mask i32[128,2000]
//  4 anco_text_e (dead)             5 label (dead)
//  6..9   text_adp (dead code in reference)
// 10..13  audio_adp ln_g(384)/ln_b(384)/W(384x128)/b(128)
// 14..29  enc: Wq bq Wk bk Wv bv Wo bo ln1_g ln1_b W1 b1 W2 b2 ln2_g ln2_b
// 30..35  gru: W_ih(300x100) W_hh(300x100) b_ih b_hh fc_W(100x100) fc_b(100)
// 36 dense_W(100)  37 dense_b(1)
//
// All dense GEMMs (adapter proj, sim, QKV, Wo, FFN1, FFN2, GRU input gates)
// run on v_wmma_f32_16x16x32_bf16. Only LN/softmax/gates/conv are VALU f32.
// ---------------------------------------------------------------------------

typedef __attribute__((ext_vector_type(16))) __bf16 v16bf;
typedef __attribute__((ext_vector_type(8)))  float  v8f;

struct U128 { unsigned int x, y, z, w; };
union FragBF { U128 u[2]; v16bf v; };   // two 16B chunks -> one 16x bf16 fragment

#define LN_EPS 1e-5f

__device__ __forceinline__ float sigm_(float x) { return 1.f / (1.f + __expf(-x)); }
__device__ __forceinline__ float silu_(float x) { return x / (1.f + __expf(-x)); }

// ---------------------------------------------------------------------------
// K0: prep — transpose audio W (384x128 f32) -> Wt bf16[128][384];
//     fold fc_W @ dense_W into w_comb[100], bias into slot 100.
// ---------------------------------------------------------------------------
__global__ void k_prep(const float* __restrict__ W, const float* __restrict__ fcW,
                       const float* __restrict__ fcb, const float* __restrict__ dW,
                       const float* __restrict__ db,
                       __bf16* __restrict__ Wt, float* __restrict__ wcomb) {
  int tid = threadIdx.x;
  for (int i = tid; i < 384 * 128; i += blockDim.x) {
    int n = i / 384, k = i % 384;
    Wt[i] = (__bf16)W[k * 128 + n];
  }
  if (tid < 100) {
    float s = 0.f;
    for (int j = 0; j < 100; ++j) s += fcW[tid * 100 + j] * dW[j];
    wcomb[tid] = s;
  }
  if (tid == 0) {
    float s = 0.f;
    for (int j = 0; j < 100; ++j) s += fcb[j] * dW[j];
    wcomb[100] = s + db[0];
  }
}

// ---------------------------------------------------------------------------
// Generic weight transpose+pad to bf16 WMMA-B layout Wt[n][k].
// src_nk=0: W is (K x N) row-major -> Wt[n][k] = W[k*N+n]
// src_nk=1: W is (N x K) row-major -> Wt[n][k] = W[n*K+k]   (used for W_ih)
// ---------------------------------------------------------------------------
__global__ void k_tpad(const float* __restrict__ W, int K, int N,
                       __bf16* __restrict__ Wt, int Nalloc, int Kpad, int src_nk) {
  int total = Nalloc * Kpad;
  for (int e = blockIdx.x * blockDim.x + threadIdx.x; e < total;
       e += gridDim.x * blockDim.x) {
    int n = e / Kpad, k = e % Kpad;
    float v = 0.f;
    if (n < N && k < K) v = src_nk ? W[n * K + k] : W[k * N + n];
    Wt[e] = (__bf16)v;
  }
}

// Stage activations f32[B][100][K] -> bf16[B][112][Kpad], zero-padded.
__global__ void k_cvt(const float* __restrict__ src, int K, int Kpad,
                      __bf16* __restrict__ dst) {
  int b = blockIdx.x, tid = threadIdx.x;
  int total = 112 * Kpad;
  for (int e = tid; e < total; e += blockDim.x) {
    int j = e / Kpad, k = e % Kpad;
    float v = (j < 100 && k < K) ? src[(size_t)b * 100 * K + j * K + k] : 0.f;
    dst[(size_t)b * total + e] = (__bf16)v;
  }
}

// ---------------------------------------------------------------------------
// Generic WMMA GEMM: out[b][row<100][col<N] = act(X(112xKpad) @ Wt^T + bias).
// Block = 128 thr (4 waves): 7 M-tiles x 2 N-tiles; grid=(ceil(Ntiles/2), B).
// act: 0=none, 1=relu.
// ---------------------------------------------------------------------------
__global__ void k_gemm(const __bf16* __restrict__ X, int Kpad,
                       const __bf16* __restrict__ Wt, int Nalloc,
                       const float* __restrict__ bias, int N,
                       float* __restrict__ out, int act) {
  int b = blockIdx.y, pair = blockIdx.x, tid = threadIdx.x;
  int lane = tid & 31, wave = tid >> 5;
  int half = lane >> 4, mn = lane & 15;
  const __bf16* Xb = X + (size_t)b * 112 * Kpad;
  int kiter = Kpad >> 5;
  for (int t = wave; t < 14; t += 4) {
    int mt = t % 7, ntl = t / 7;
    int ntile = pair * 2 + ntl;
    if (ntile * 16 >= Nalloc) continue;            // wave-uniform
    v8f acc = {0.f, 0.f, 0.f, 0.f, 0.f, 0.f, 0.f, 0.f};
    const __bf16* arow = Xb + (size_t)(mt * 16 + mn) * Kpad;
    const __bf16* brow = Wt + (size_t)(ntile * 16 + mn) * Kpad;
    for (int kt = 0; kt < kiter; ++kt) {
      int k0 = kt * 32 + half * 8, k1 = k0 + 16;
      FragBF af, bf;
      af.u[0] = *(const U128*)&arow[k0]; af.u[1] = *(const U128*)&arow[k1];
      bf.u[0] = *(const U128*)&brow[k0]; bf.u[1] = *(const U128*)&brow[k1];
      acc = __builtin_amdgcn_wmma_f32_16x16x32_bf16(false, af.v, false, bf.v, (short)0, acc, false, false);
    }
    int col = ntile * 16 + mn;
    if (col < N) {
      float bs = bias ? bias[col] : 0.f;
      #pragma unroll
      for (int r = 0; r < 8; ++r) {
        int row = mt * 16 + r + 8 * half;          // C/D layout: lanes16-31 -> M+8
        if (row < 100) {
          float v = acc[r] + bs;
          if (act == 1) v = fmaxf(v, 0.f);
          out[(size_t)b * 100 * N + (size_t)row * N + col] = v;
        }
      }
    }
  }
}

// ---------------------------------------------------------------------------
// K1: per-batch mask lengths + bilinear row/col indices & weights.
// ---------------------------------------------------------------------------
__global__ void k_index(const int* __restrict__ amask, const int* __restrict__ cmask,
                        int* __restrict__ r0, int* __restrict__ r1, float* __restrict__ wr,
                        int* __restrict__ c0, int* __restrict__ c1, float* __restrict__ wc) {
  __shared__ int red[128];
  __shared__ int sLa, sLc;
  int b = blockIdx.x, tid = threadIdx.x;
  int s = 0;
  for (int i = tid; i < 100; i += 128) s += amask[b * 100 + i];
  red[tid] = s; __syncthreads();
  for (int o = 64; o > 0; o >>= 1) { if (tid < o) red[tid] += red[tid + o]; __syncthreads(); }
  if (tid == 0) sLa = red[0];
  __syncthreads();
  s = 0;
  for (int i = tid; i < 2000; i += 128) s += cmask[b * 2000 + i];
  red[tid] = s; __syncthreads();
  for (int o = 64; o > 0; o >>= 1) { if (tid < o) red[tid] += red[tid + o]; __syncthreads(); }
  if (tid == 0) sLc = red[0];
  __syncthreads();
  int La = sLa, Lc = sLc;
  if (tid < 100) {
    float sr = (float)tid * (float)(La - 1) / 99.f;
    int a0 = (int)floorf(sr);
    r0[b * 100 + tid] = a0;
    r1[b * 100 + tid] = min(a0 + 1, La - 1);
    wr[b * 100 + tid] = sr - (float)a0;
    float sc = (float)tid * (float)(Lc - 1) / 99.f;
    int b0 = (int)floorf(sc);
    c0[b * 100 + tid] = b0;
    c1[b * 100 + tid] = min(b0 + 1, Lc - 1);
    wc[b * 100 + tid] = sc - (float)b0;
  }
}

// ---------------------------------------------------------------------------
// K2/K3: fused LayerNorm(384) -> GEMM(384->128, bf16 WMMA) -> SiLU -> L2 norm.
// mode 0: rows from anco_speech (nvalid=100, dst stride 112)
// mode 1: rows gathered from com_speech via c0/c1 (nvalid=200, dst stride 208)
// ---------------------------------------------------------------------------
__global__ void k_proj(const float* __restrict__ srcA, const float* __restrict__ srcC,
                       const int* __restrict__ c0, const int* __restrict__ c1,
                       const float* __restrict__ ln_g, const float* __restrict__ ln_b,
                       const __bf16* __restrict__ Wt, const float* __restrict__ bias,
                       __bf16* __restrict__ dst, int dst_stride, int nvalid, int mode) {
  __shared__ __bf16 At[16 * 400];
  __shared__ float outT[16 * 128];
  __shared__ float red[16][8];
  __shared__ float red2[16][8];
  __shared__ float stat[16][2];
  __shared__ const float* rowptr[16];

  int b = blockIdx.y, tile = blockIdx.x, tid = threadIdx.x;
  int lr = tid >> 3, sub = tid & 7;
  int grow = tile * 16 + lr;

  if (sub == 0) {
    const float* p = nullptr;
    if (grow < nvalid) {
      if (mode == 0) {
        p = srcA + ((size_t)b * 100 + grow) * 384;
      } else {
        int col = (grow < 100) ? c0[b * 100 + grow] : c1[b * 100 + (grow - 100)];
        p = srcC + ((size_t)b * 2000 + col) * 384;
      }
    }
    rowptr[lr] = p;
  }
  __syncthreads();

  const float* p = rowptr[lr];
  int kbase = sub * 48;
  float sum = 0.f, sq = 0.f;
  if (p) {
    #pragma unroll 4
    for (int k = 0; k < 48; ++k) { float v = p[kbase + k]; sum += v; sq += v * v; }
  }
  red[lr][sub] = sum; red2[lr][sub] = sq;
  __syncthreads();
  if (sub == 0) {
    float s0 = 0.f, q0 = 0.f;
    for (int i = 0; i < 8; ++i) { s0 += red[lr][i]; q0 += red2[lr][i]; }
    float m = s0 * (1.f / 384.f);
    float var = q0 * (1.f / 384.f) - m * m;
    stat[lr][0] = m; stat[lr][1] = rsqrtf(var + LN_EPS);
  }
  __syncthreads();
  {
    float m = stat[lr][0], inv = stat[lr][1];
    for (int k = 0; k < 48; ++k) {
      int kk = kbase + k;
      float v = p ? ((p[kk] - m) * inv * ln_g[kk] + ln_b[kk]) : 0.f;
      At[lr * 400 + kk] = (__bf16)v;
    }
  }
  __syncthreads();

  int lane = tid & 31, wave = tid >> 5;
  int half = lane >> 4, mn = lane & 15;
  v8f acc0 = {0.f, 0.f, 0.f, 0.f, 0.f, 0.f, 0.f, 0.f};
  v8f acc1 = acc0;
  int col0 = (2 * wave) * 16 + mn;
  int col1 = (2 * wave + 1) * 16 + mn;
  for (int kt = 0; kt < 12; ++kt) {
    int k0 = kt * 32 + half * 8;
    int k1 = k0 + 16;
    FragBF a, b0, b1;
    a.u[0]  = *(const U128*)&At[mn * 400 + k0];
    a.u[1]  = *(const U128*)&At[mn * 400 + k1];
    b0.u[0] = *(const U128*)&Wt[(size_t)col0 * 384 + k0];
    b0.u[1] = *(const U128*)&Wt[(size_t)col0 * 384 + k1];
    b1.u[0] = *(const U128*)&Wt[(size_t)col1 * 384 + k0];
    b1.u[1] = *(const U128*)&Wt[(size_t)col1 * 384 + k1];
    acc0 = __builtin_amdgcn_wmma_f32_16x16x32_bf16(false, a.v, false, b0.v, (short)0, acc0, false, false);
    acc1 = __builtin_amdgcn_wmma_f32_16x16x32_bf16(false, a.v, false, b1.v, (short)0, acc1, false, false);
  }
  #pragma unroll
  for (int r = 0; r < 8; ++r) {
    int m = r + 8 * half;
    outT[m * 128 + col0] = silu_(acc0[r] + bias[col0]);
    outT[m * 128 + col1] = silu_(acc1[r] + bias[col1]);
  }
  __syncthreads();

  {
    float q0 = 0.f;
    for (int c = 0; c < 16; ++c) { float v = outT[lr * 128 + sub * 16 + c]; q0 += v * v; }
    red[lr][sub] = q0;
  }
  __syncthreads();
  if (sub == 0) {
    float q0 = 0.f;
    for (int i = 0; i < 8; ++i) q0 += red[lr][i];
    stat[lr][0] = 1.f / (sqrtf(q0) + 1e-8f);
  }
  __syncthreads();
  {
    float invn = stat[lr][0];
    bool valid = (grow < nvalid);
    __bf16* o = dst + ((size_t)b * dst_stride + grow) * 128 + sub * 16;
    for (int c = 0; c < 16; ++c) {
      float v = valid ? outT[lr * 128 + sub * 16 + c] * invn : 0.f;
      o[c] = (__bf16)v;
    }
  }
}

// ---------------------------------------------------------------------------
// Bilinear folded into vector blends: dst[j] = (1-w)src[i0] + w*src[i1].
// ---------------------------------------------------------------------------
__global__ void k_blend(const __bf16* __restrict__ src, int src_stride,
                        const int* __restrict__ i0, const int* __restrict__ i1,
                        const float* __restrict__ w, __bf16* __restrict__ dst) {
  int b = blockIdx.x, tid = threadIdx.x;
  for (int e = tid; e < 112 * 128; e += blockDim.x) {
    int j = e >> 7, cc = e & 127;
    float v = 0.f;
    if (j < 100) {
      int a0 = i0 ? i0[b * 100 + j] : j;
      int a1 = i1 ? i1[b * 100 + j] : (100 + j);
      float ww = w[b * 100 + j];
      float v0 = (float)src[((size_t)b * src_stride + a0) * 128 + cc];
      float v1 = (float)src[((size_t)b * src_stride + a1) * 128 + cc];
      v = v0 + ww * (v1 - v0);
    }
    dst[((size_t)b * 112 + j) * 128 + cc] = (__bf16)v;
  }
}

// ---------------------------------------------------------------------------
// up = ab @ cb^T (112x112, K=128) via WMMA into LDS, then 5x5 Gaussian blur.
// ---------------------------------------------------------------------------
__global__ void k_simconv(const __bf16* __restrict__ ab, const __bf16* __restrict__ cb,
                          float* __restrict__ sm) {
  __shared__ float up[112 * 112];
  int b = blockIdx.x, tid = threadIdx.x;
  int lane = tid & 31, wave = tid >> 5;
  int half = lane >> 4, mn = lane & 15;
  const __bf16* A = ab + (size_t)b * 112 * 128;
  const __bf16* B = cb + (size_t)b * 112 * 128;
  for (int t = wave; t < 49; t += 4) {
    int mt = t / 7, nt = t % 7;
    v8f acc = {0.f, 0.f, 0.f, 0.f, 0.f, 0.f, 0.f, 0.f};
    const __bf16* arow = A + (size_t)(mt * 16 + mn) * 128;
    const __bf16* brow = B + (size_t)(nt * 16 + mn) * 128;
    #pragma unroll
    for (int kt = 0; kt < 4; ++kt) {
      int k0 = kt * 32 + half * 8, k1 = k0 + 16;
      FragBF af, bf;
      af.u[0] = *(const U128*)&arow[k0]; af.u[1] = *(const U128*)&arow[k1];
      bf.u[0] = *(const U128*)&brow[k0]; bf.u[1] = *(const U128*)&brow[k1];
      acc = __builtin_amdgcn_wmma_f32_16x16x32_bf16(false, af.v, false, bf.v, (short)0, acc, false, false);
    }
    #pragma unroll
    for (int r = 0; r < 8; ++r)
      up[(mt * 16 + r + 8 * half) * 112 + nt * 16 + mn] = acc[r];
  }
  __syncthreads();
  const float g1[5] = {0.054488685f, 0.244201342f, 0.402619946f, 0.244201342f, 0.054488685f};
  for (int e = tid; e < 10000; e += 128) {
    int j = e / 100, kk = e % 100;
    float s = 0.f;
    #pragma unroll
    for (int dy = -2; dy <= 2; ++dy) {
      int jj = j + dy;
      if ((unsigned)jj >= 100u) continue;
      float gy = g1[dy + 2];
      #pragma unroll
      for (int dx = -2; dx <= 2; ++dx) {
        int kx = kk + dx;
        if ((unsigned)kx >= 100u) continue;
        s += gy * g1[dx + 2] * up[jj * 112 + kx];
      }
    }
    sm[((size_t)b * 100 + j) * 100 + kk] = s;
  }
}

// ---------------------------------------------------------------------------
// Attention core (dh=25 -> f32 VALU, LDS-staged). One block / (batch, head).
// ---------------------------------------------------------------------------
__global__ void k_attn(const float* __restrict__ q, const float* __restrict__ k,
                       const float* __restrict__ v, float* __restrict__ o) {
  __shared__ float qh[100][26], kh[100][26], vh[100][26];
  int bh = blockIdx.x, b = bh >> 2, h = bh & 3;
  int tid = threadIdx.x;
  const float* qb = q + (size_t)b * 10000 + h * 25;
  const float* kb = k + (size_t)b * 10000 + h * 25;
  const float* vb = v + (size_t)b * 10000 + h * 25;
  for (int e = tid; e < 2500; e += blockDim.x) {
    int t = e / 25, d = e % 25;
    qh[t][d] = qb[t * 100 + d];
    kh[t][d] = kb[t * 100 + d];
    vh[t][d] = vb[t * 100 + d];
  }
  __syncthreads();
  if (tid < 100) {
    int i = tid;
    const float scale = 0.2f;  // 1/sqrt(25)
    float mx = -1e30f;
    for (int j = 0; j < 100; ++j) {
      float s = 0.f;
      for (int d = 0; d < 25; ++d) s += qh[i][d] * kh[j][d];
      mx = fmaxf(mx, s * scale);
    }
    float se = 0.f, acc[25];
    #pragma unroll
    for (int d = 0; d < 25; ++d) acc[d] = 0.f;
    for (int j = 0; j < 100; ++j) {
      float s = 0.f;
      for (int d = 0; d < 25; ++d) s += qh[i][d] * kh[j][d];
      float e2 = __expf(s * scale - mx);
      se += e2;
      for (int d = 0; d < 25; ++d) acc[d] += e2 * vh[j][d];
    }
    float inv = 1.f / se;
    float* ob = o + (size_t)b * 10000 + i * 100 + h * 25;
    for (int d = 0; d < 25; ++d) ob[d] = acc[d] * inv;
  }
}

// out = LN(y + res) row-wise over 100 features.
__global__ void k_res_ln(const float* __restrict__ y, const float* __restrict__ res,
                         const float* __restrict__ g, const float* __restrict__ be,
                         float* __restrict__ out) {
  int b = blockIdx.x, tid = threadIdx.x;
  if (tid < 100) {
    const float* yr = y + (size_t)b * 10000 + tid * 100;
    const float* rr = res + (size_t)b * 10000 + tid * 100;
    float s = 0.f, q = 0.f;
    for (int n = 0; n < 100; ++n) { float v = yr[n] + rr[n]; s += v; q += v * v; }
    float m = s * 0.01f;
    float inv = rsqrtf(q * 0.01f - m * m + LN_EPS);
    float* ob = out + (size_t)b * 10000 + tid * 100;
    for (int n = 0; n < 100; ++n) ob[n] = ((yr[n] + rr[n]) - m) * inv * g[n] + be[n];
  }
}

// ---------------------------------------------------------------------------
// GRU: input gates gi precomputed by WMMA GEMM; only h@W_hh^T is sequential.
// Folded fc+dense head via wcomb.
// ---------------------------------------------------------------------------
__global__ void k_gru(const float* __restrict__ giall, const float* __restrict__ Whh,
                      const float* __restrict__ bhh, const float* __restrict__ wcomb,
                      float* __restrict__ out) {
  __shared__ float hS[100], gh[300], red[128];
  int b = blockIdx.x, tid = threadIdx.x;
  if (tid < 100) hS[tid] = 0.f;
  __syncthreads();
  for (int t = 0; t < 100; ++t) {
    for (int g = tid; g < 300; g += 128) {
      float c = bhh[g];
      const float* wh = Whh + (size_t)g * 100;
      for (int kk = 0; kk < 100; ++kk) c += hS[kk] * wh[kk];
      gh[g] = c;
    }
    __syncthreads();
    if (tid < 100) {
      const float* gi = giall + (size_t)b * 30000 + (size_t)t * 300;
      float r = sigm_(gi[tid] + gh[tid]);
      float z = sigm_(gi[100 + tid] + gh[100 + tid]);
      float n = tanhf(gi[200 + tid] + r * gh[200 + tid]);
      hS[tid] = (1.f - z) * n + z * hS[tid];
    }
    __syncthreads();
  }
  float s = (tid < 100) ? hS[tid] * wcomb[tid] : 0.f;
  red[tid] = s;
  __syncthreads();
  for (int o = 64; o > 0; o >>= 1) { if (tid < o) red[tid] += red[tid + o]; __syncthreads(); }
  if (tid == 0) out[b] = red[0] + wcomb[100];
}

// ---------------------------------------------------------------------------
extern "C" void kernel_launch(void* const* d_in, const int* in_sizes, int n_in,
                              void* d_out, int out_size, void* d_ws, size_t ws_size,
                              hipStream_t stream) {
  if (n_in < 38) return;
  const int B = 128;
  const float* anco  = (const float*)d_in[0];
  const int*   amask = (const int*)d_in[1];
  const float* com   = (const float*)d_in[2];
  const int*   cmask = (const int*)d_in[3];
  const float* aud_g = (const float*)d_in[10];
  const float* aud_b = (const float*)d_in[11];
  const float* aud_W = (const float*)d_in[12];
  const float* aud_bias = (const float*)d_in[13];
  const float* Wq = (const float*)d_in[14]; const float* bq = (const float*)d_in[15];
  const float* Wk = (const float*)d_in[16]; const float* bk = (const float*)d_in[17];
  const float* Wv = (const float*)d_in[18]; const float* bv = (const float*)d_in[19];
  const float* Wo = (const float*)d_in[20]; const float* bo = (const float*)d_in[21];
  const float* ln1g = (const float*)d_in[22]; const float* ln1b = (const float*)d_in[23];
  const float* W1 = (const float*)d_in[24]; const float* b1 = (const float*)d_in[25];
  const float* W2 = (const float*)d_in[26]; const float* b2 = (const float*)d_in[27];
  const float* ln2g = (const float*)d_in[28]; const float* ln2b = (const float*)d_in[29];
  const float* Wih = (const float*)d_in[30]; const float* Whh = (const float*)d_in[31];
  const float* bih = (const float*)d_in[32]; const float* bhh = (const float*)d_in[33];
  const float* fcW = (const float*)d_in[34]; const float* fcb = (const float*)d_in[35];
  const float* dW  = (const float*)d_in[36]; const float* db  = (const float*)d_in[37];
  float* out = (float*)d_out;

  // ---- workspace carve-out (~102 MB) ----
  char* w = (char*)d_ws;
  size_t off = 0;
  auto alloc = [&](size_t bytes) -> void* {
    void* p = w + off;
    off += (bytes + 255) & ~(size_t)255;
    return p;
  };
  __bf16* Wt   = (__bf16*)alloc((size_t)128 * 384 * 2);
  float* wcomb = (float*)alloc(101 * 4);
  int* r0 = (int*)alloc((size_t)B * 100 * 4);
  int* r1 = (int*)alloc((size_t)B * 100 * 4);
  float* wr = (float*)alloc((size_t)B * 100 * 4);
  int* c0 = (int*)alloc((size_t)B * 100 * 4);
  int* c1 = (int*)alloc((size_t)B * 100 * 4);
  float* wc = (float*)alloc((size_t)B * 100 * 4);
  __bf16* a_n   = (__bf16*)alloc((size_t)B * 112 * 128 * 2);
  __bf16* c_sel = (__bf16*)alloc((size_t)B * 208 * 128 * 2);
  __bf16* abv   = (__bf16*)alloc((size_t)B * 112 * 128 * 2);
  __bf16* cbv   = (__bf16*)alloc((size_t)B * 112 * 128 * 2);
  float* sm   = (float*)alloc((size_t)B * 10000 * 4);
  float* qb_  = (float*)alloc((size_t)B * 10000 * 4);
  float* kb_  = (float*)alloc((size_t)B * 10000 * 4);
  float* vb_  = (float*)alloc((size_t)B * 10000 * 4);
  float* ao   = (float*)alloc((size_t)B * 10000 * 4);
  float* x1   = (float*)alloc((size_t)B * 10000 * 4);
  float* h400 = (float*)alloc((size_t)B * 40000 * 4);
  float* x2   = (float*)alloc((size_t)B * 10000 * 4);
  float* giall = (float*)alloc((size_t)B * 30000 * 4);
  __bf16* xbf  = (__bf16*)alloc((size_t)B * 112 * 416 * 2);   // staging, max Kpad=416
  // encoder / GRU weights in WMMA-B bf16 layout
  __bf16* Wtq  = (__bf16*)alloc((size_t)112 * 128 * 2);
  __bf16* Wtk  = (__bf16*)alloc((size_t)112 * 128 * 2);
  __bf16* Wtv  = (__bf16*)alloc((size_t)112 * 128 * 2);
  __bf16* Wto  = (__bf16*)alloc((size_t)112 * 128 * 2);
  __bf16* Wt1  = (__bf16*)alloc((size_t)400 * 128 * 2);
  __bf16* Wt2  = (__bf16*)alloc((size_t)112 * 416 * 2);
  __bf16* Wtih = (__bf16*)alloc((size_t)304 * 128 * 2);
  (void)ws_size; (void)in_sizes; (void)out_size;

  // ---- weight prep ----
  k_prep<<<1, 256, 0, stream>>>(aud_W, fcW, fcb, dW, db, Wt, wcomb);
  k_tpad<<<32, 256, 0, stream>>>(Wq, 100, 100, Wtq, 112, 128, 0);
  k_tpad<<<32, 256, 0, stream>>>(Wk, 100, 100, Wtk, 112, 128, 0);
  k_tpad<<<32, 256, 0, stream>>>(Wv, 100, 100, Wtv, 112, 128, 0);
  k_tpad<<<32, 256, 0, stream>>>(Wo, 100, 100, Wto, 112, 128, 0);
  k_tpad<<<32, 256, 0, stream>>>(W1, 100, 400, Wt1, 400, 128, 0);
  k_tpad<<<32, 256, 0, stream>>>(W2, 400, 100, Wt2, 112, 416, 0);
  k_tpad<<<32, 256, 0, stream>>>(Wih, 100, 300, Wtih, 304, 128, 1);

  // ---- adapter + similarity (WMMA, gathers only needed columns) ----
  k_index<<<B, 128, 0, stream>>>(amask, cmask, r0, r1, wr, c0, c1, wc);
  k_proj<<<dim3(7, B), 128, 0, stream>>>(anco, com, c0, c1, aud_g, aud_b, Wt, aud_bias,
                                         a_n, 112, 100, 0);
  k_proj<<<dim3(13, B), 128, 0, stream>>>(anco, com, c0, c1, aud_g, aud_b, Wt, aud_bias,
                                          c_sel, 208, 200, 1);
  k_blend<<<B, 256, 0, stream>>>(a_n, 112, r0, r1, wr, abv);
  k_blend<<<B, 256, 0, stream>>>(c_sel, 208, nullptr, nullptr, wc, cbv);
  k_simconv<<<B, 128, 0, stream>>>(abv, cbv, sm);

  // ---- encoder: all GEMMs via WMMA ----
  k_cvt<<<B, 256, 0, stream>>>(sm, 100, 128, xbf);
  k_gemm<<<dim3(4, B), 128, 0, stream>>>(xbf, 128, Wtq, 112, bq, 100, qb_, 0);
  k_gemm<<<dim3(4, B), 128, 0, stream>>>(xbf, 128, Wtk, 112, bk, 100, kb_, 0);
  k_gemm<<<dim3(4, B), 128, 0, stream>>>(xbf, 128, Wtv, 112, bv, 100, vb_, 0);
  k_attn<<<B * 4, 128, 0, stream>>>(qb_, kb_, vb_, ao);
  k_cvt<<<B, 256, 0, stream>>>(ao, 100, 128, xbf);
  k_gemm<<<dim3(4, B), 128, 0, stream>>>(xbf, 128, Wto, 112, bo, 100, qb_, 0);  // reuse qb_
  k_res_ln<<<B, 128, 0, stream>>>(qb_, sm, ln1g, ln1b, x1);
  k_cvt<<<B, 256, 0, stream>>>(x1, 100, 128, xbf);
  k_gemm<<<dim3(13, B), 128, 0, stream>>>(xbf, 128, Wt1, 400, b1, 400, h400, 1); // relu
  k_cvt<<<B, 256, 0, stream>>>(h400, 400, 416, xbf);
  k_gemm<<<dim3(4, B), 128, 0, stream>>>(xbf, 416, Wt2, 112, b2, 100, kb_, 0);  // reuse kb_
  k_res_ln<<<B, 128, 0, stream>>>(kb_, x1, ln2g, ln2b, x2);

  // ---- GRU: input gates batched through WMMA, recurrent half sequential ----
  k_cvt<<<B, 256, 0, stream>>>(x2, 100, 128, xbf);
  k_gemm<<<dim3(10, B), 128, 0, stream>>>(xbf, 128, Wtih, 304, bih, 300, giall, 0);
  k_gru<<<B, 128, 0, stream>>>(giall, Whh, bhh, wcomb, out);
}
